// CausalTimeSeriesMamba_89764816486405
// MI455X (gfx1250) — compile-verified, matrix operand order checked by reading, not run on hardware
//
#include <hip/hip_runtime.h>
#include <cstdint>
#include <cstddef>

// ---- problem dims ----
#define B_  4
#define L_  4096
#define FD  20
#define DM  512
#define NL  4
#define DS  16
#define DC  4
#define DI  1024
#define DFF 2048
#define DTR 32
#define M_  (B_ * L_)   // 16384 rows through every GEMM

typedef __attribute__((ext_vector_type(16))) __bf16 v16bf;
typedef __attribute__((ext_vector_type(8)))  __bf16 v8bf;
typedef __attribute__((ext_vector_type(8)))  float  v8f;
typedef __attribute__((ext_vector_type(4)))  int    v4i;

__device__ __forceinline__ float siluf(float x) { return x / (1.0f + __expf(-x)); }
__device__ __forceinline__ float geluf(float x) { return 0.5f * x * (1.0f + erff(x * 0.70710678118654752f)); }
__device__ __forceinline__ float softplusf(float x) { return (x > 20.0f) ? x : log1pf(__expf(x)); }

// ---- gfx1250 async global->LDS staging (ASYNCcnt path), with safe fallback ----
#if defined(__HIP_DEVICE_COMPILE__) && \
    __has_builtin(__builtin_amdgcn_global_load_async_to_lds_b128) && \
    __has_builtin(__builtin_amdgcn_s_wait_asynccnt)
#define HAVE_ASYNC_LDS 1
#else
#define HAVE_ASYNC_LDS 0
#endif

#if HAVE_ASYNC_LDS
typedef __attribute__((address_space(1))) v4i* as1_v4i_ptr;   // global b128 ptr
typedef __attribute__((address_space(3))) v4i* as3_v4i_ptr;   // LDS b128 ptr
#endif

__device__ __forceinline__ void stage_chunk(const __bf16* __restrict__ g, __bf16* l) {
#if HAVE_ASYNC_LDS
  __builtin_amdgcn_global_load_async_to_lds_b128((as1_v4i_ptr)g, (as3_v4i_ptr)l, 0, 0);
#else
  *(v8bf*)l = *(const v8bf*)g;   // global_load_b128 + ds_store_b128
#endif
}

__device__ __forceinline__ void wait_stage() {
#if HAVE_ASYNC_LDS
  __builtin_amdgcn_s_wait_asynccnt(0);
#endif
}

// ------------------------------------------------------------------
// bf16 WMMA GEMM:  C(MxN) = act(A(MxK) * W(NxK)^T + bias) [+Cold]
// Block: 256 threads = 8 waves. Tile 128(M) x 64(N), K-step 32,
// double-buffered LDS. Wave w owns rows [w*16, w*16+16) x all 64 cols
// -> 4 x v_wmma_f32_16x16x32_bf16 per wave per K-step.
// ------------------------------------------------------------------
__global__ __launch_bounds__(256) void k_gemm_wmma(
    const __bf16* __restrict__ A, const __bf16* __restrict__ W,
    const float* __restrict__ bias, float* __restrict__ Cf,
    __bf16* __restrict__ Cbf, int Mdim, int Ndim, int Kdim,
    int act, int addC)
{
  (void)Mdim;
  constexpr int LDSS  = 40;            // bf16 elems per LDS row (bank-spread pad)
  constexpr int ATILE = 128 * LDSS;
  constexpr int WTILE = 64 * LDSS;
  __shared__ __bf16 As[2 * ATILE];
  __shared__ __bf16 Ws[2 * WTILE];

  const int tid   = threadIdx.x;
  const int lane  = tid & 31;
  const int w     = tid >> 5;          // wave 0..7 -> 16-row slab
  const int tileM = blockIdx.y * 128;
  const int tileN = blockIdx.x * 64;

  v8f acc[4];
#pragma unroll
  for (int b = 0; b < 4; ++b)
#pragma unroll
    for (int j = 0; j < 8; ++j) acc[b][j] = 0.0f;

  // staging: one 128-bit chunk per (row, k-quarter); thread -> chunk map
  const int cr = tid >> 2;             // 0..63
  const int ck = (tid & 3) * 8;        // 0,8,16,24

  auto stageTile = [&](int k0, int buf) {
    __bf16* Ab = &As[buf * ATILE];
    __bf16* Wb = &Ws[buf * WTILE];
#pragma unroll
    for (int s = 0; s < 2; ++s) {      // A: 128 rows (two 64-row halves)
      const int r = cr + s * 64;
      stage_chunk(A + (size_t)(tileM + r) * Kdim + k0 + ck, Ab + r * LDSS + ck);
    }
    const int n = tileN + cr;          // W: 64 rows, zero-pad OOB (head N=20)
    if (n < Ndim) {
      stage_chunk(W + (size_t)n * Kdim + k0 + ck, Wb + cr * LDSS + ck);
    } else {
      v8bf zv;
#pragma unroll
      for (int i = 0; i < 8; ++i) zv[i] = (__bf16)0.0f;
      *(v8bf*)(Wb + cr * LDSS + ck) = zv;
    }
  };

  stageTile(0, 0);
  wait_stage();
  __syncthreads();

  const int nsteps = Kdim >> 5;
  for (int s = 0; s < nsteps; ++s) {
    const int buf = s & 1;
    if (s + 1 < nsteps) stageTile((s + 1) * 32, buf ^ 1);   // overlap with math

    const __bf16* Ab = &As[buf * ATILE];
    const __bf16* Wb = &Ws[buf * WTILE];
    // 16-bit fragment layout: lanes 0-15 K={0..7,16..23}; lanes 16-31 K={8..15,24..31}
    const int half = (lane >> 4) * 8;
    const __bf16* ap = &Ab[(w * 16 + (lane & 15)) * LDSS];
    v16bf af;
#pragma unroll
    for (int i = 0; i < 8; ++i) { af[i] = ap[half + i]; af[i + 8] = ap[16 + half + i]; }
    v16bf bfr[4];
#pragma unroll
    for (int b = 0; b < 4; ++b) {
      const __bf16* bp = &Wb[(b * 16 + (lane & 15)) * LDSS];
#pragma unroll
      for (int i = 0; i < 8; ++i) { bfr[b][i] = bp[half + i]; bfr[b][i + 8] = bp[16 + half + i]; }
    }
#pragma unroll
    for (int b = 0; b < 4; ++b)
      acc[b] = __builtin_amdgcn_wmma_f32_16x16x32_bf16(false, af, false, bfr[b],
                                                       (short)0, acc[b], false, false);
    wait_stage();
    __syncthreads();
  }

  // Epilogue. C layout: VGPR j -> M = j (+8 for lanes 16-31), N = lane&15.
  const int rowBase = tileM + w * 16 + ((lane >> 4) << 3);
  const int colBase = tileN + (lane & 15);
#pragma unroll
  for (int b = 0; b < 4; ++b) {
    const int col = colBase + b * 16;
    if (col >= Ndim) continue;
    const float bv = bias ? bias[col] : 0.0f;
#pragma unroll
    for (int j = 0; j < 8; ++j) {
      float v = acc[b][j] + bv;
      if (act == 1)      v = siluf(v);
      else if (act == 2) v = geluf(v);
      else if (act == 3) v = softplusf(v);
      const size_t idx = (size_t)(rowBase + j) * Ndim + col;
      if (addC) v += Cf[idx];
      if (Cf)  Cf[idx]  = v;
      if (Cbf) Cbf[idx] = (__bf16)v;
    }
  }
}

// ---- small elementwise / helper kernels ----
__global__ void k_cvt_bf16(const float* __restrict__ in, __bf16* __restrict__ out, int n) {
  int i = blockIdx.x * blockDim.x + threadIdx.x;
  if (i < n) out[i] = (__bf16)in[i];
}

__global__ void k_negexp(const float* __restrict__ in, float* __restrict__ out, int n) {
  int i = blockIdx.x * blockDim.x + threadIdx.x;
  if (i < n) out[i] = -__expf(in[i]);
}

// h = x @ W_in^T + b_in   (K = 20 too small for WMMA; cheap VALU kernel)
__global__ void k_in_proj(const float* __restrict__ x, const float* __restrict__ Wf,
                          const float* __restrict__ bf_, float* __restrict__ h) {
  int i = blockIdx.x * blockDim.x + threadIdx.x;   // over M_*DM
  int m = i / DM, d = i % DM;
  const float* xr = x + (size_t)m * FD;
  const float* wr = Wf + (size_t)d * FD;
  float acc = bf_[d];
#pragma unroll
  for (int k = 0; k < FD; ++k) acc = fmaf(xr[k], wr[k], acc);
  h[(size_t)m * DM + d] = acc;
}

// row LayerNorm (DM=512) -> bf16 output (feeds WMMA GEMMs)
__global__ __launch_bounds__(256) void k_layernorm_bf16(
    const float* __restrict__ h, const float* __restrict__ w,
    const float* __restrict__ b, __bf16* __restrict__ out) {
  __shared__ float red[16];
  const int m = blockIdx.x;
  const int t = threadIdx.x;
  const float* row = h + (size_t)m * DM;
  float x0 = row[t], x1 = row[t + 256];
  float s = x0 + x1, ss = x0 * x0 + x1 * x1;
#pragma unroll
  for (int o = 16; o > 0; o >>= 1) {
    s  += __shfl_down(s,  o, 32);
    ss += __shfl_down(ss, o, 32);
  }
  const int lane = t & 31, wv = t >> 5;
  if (lane == 0) { red[wv] = s; red[8 + wv] = ss; }
  __syncthreads();
  if (t == 0) {
    float S = 0.f, SS = 0.f;
    for (int i = 0; i < 8; ++i) { S += red[i]; SS += red[8 + i]; }
    red[0] = S; red[8] = SS;
  }
  __syncthreads();
  const float mean = red[0] * (1.0f / DM);
  const float var  = red[8] * (1.0f / DM) - mean * mean;
  const float rstd = rsqrtf(var + 1e-5f);
  out[(size_t)m * DM + t]       = (__bf16)((x0 - mean) * rstd * w[t]       + b[t]);
  out[(size_t)m * DM + t + 256] = (__bf16)((x1 - mean) * rstd * w[t + 256] + b[t + 256]);
}

// causal depthwise conv (DC=4) over bf16 u-half of xz, + bias, SiLU -> bf16
__global__ void k_conv_silu(const __bf16* __restrict__ xzbf,
                            const float* __restrict__ cw, const float* __restrict__ cb,
                            __bf16* __restrict__ ubf) {
  int i = blockIdx.x * blockDim.x + threadIdx.x;   // over M_*DI
  int m = i / DI, d = i % DI;
  int bidx = m / L_, l = m % L_;
  const float* w4 = cw + d * DC;
  float acc = cb[d];
#pragma unroll
  for (int j = 0; j < DC; ++j) {
    int ll = l - (DC - 1) + j;
    if (ll >= 0)
      acc = fmaf(w4[j], (float)xzbf[(size_t)(bidx * L_ + ll) * (2 * DI) + d], acc);
  }
  ubf[i] = (__bf16)siluf(acc);
}

__global__ void k_slice_dtr(const float* __restrict__ xdb, __bf16* __restrict__ out) {
  int i = blockIdx.x * blockDim.x + threadIdx.x;   // over M_*DTR
  int m = i / DTR, j = i % DTR;
  out[i] = (__bf16)xdb[(size_t)m * (DTR + 2 * DS) + j];
}

// Selective scan: one lane per (b, d, n); 16 lanes per channel.
// dt/u loaded once per group and shfl-broadcast; state dot via shfl_xor.
__global__ __launch_bounds__(256) void k_scan(
    const __bf16* __restrict__ dtb, const __bf16* __restrict__ u,
    const float* __restrict__ xdb, const float* __restrict__ Aneg,
    float* __restrict__ ys) {
  const int idx  = blockIdx.x * blockDim.x + threadIdx.x;  // B_*DI*DS threads
  const int lane = threadIdx.x & 31;
  const int n    = idx & 15;
  const int p    = idx >> 4;          // (b, d)
  const int d    = p % DI;
  const int bidx = p / DI;
  const int src  = lane & 16;         // group broadcast source lane (0 or 16)
  const float An = Aneg[d * DS + n];
  float hc = 0.0f;
  const size_t base = (size_t)bidx * L_;
  for (int l = 0; l < L_; ++l) {
    const size_t m = base + l;
    float dtv = 0.0f, uv = 0.0f;
    if (n == 0) {
      dtv = (float)dtb[m * DI + d];
      uv  = (float)u[m * DI + d];
    }
    dtv = __shfl(dtv, src, 32);
    uv  = __shfl(uv,  src, 32);
    const float* xr = xdb + m * (DTR + 2 * DS);
    __builtin_prefetch(xr + (DTR + 2 * DS), 0, 1);   // next row's B/C
    const float dA = __expf(dtv * An);
    hc = fmaf(dA, hc, dtv * uv * xr[DTR + n]);
    float y = hc * xr[DTR + DS + n];
    y += __shfl_xor(y, 1, 32);
    y += __shfl_xor(y, 2, 32);
    y += __shfl_xor(y, 4, 32);
    y += __shfl_xor(y, 8, 32);
    if (n == 0) ys[m * DI + d] = y;
  }
}

// y = (ys + u*D) * silu(z)  -> bf16 for out_proj GEMM
__global__ void k_gate(const float* __restrict__ ys, const __bf16* __restrict__ u,
                       const __bf16* __restrict__ xzbf, const float* __restrict__ Dp,
                       __bf16* __restrict__ ybf) {
  int i = blockIdx.x * blockDim.x + threadIdx.x;   // over M_*DI
  int m = i / DI, d = i % DI;
  float z = (float)xzbf[(size_t)m * (2 * DI) + DI + d];
  float v = (ys[i] + (float)u[i] * Dp[d]) * siluf(z);
  ybf[i] = (__bf16)v;
}

// ------------------------------------------------------------------
extern "C" void kernel_launch(void* const* d_in, const int* in_sizes, int n_in,
                              void* d_out, int out_size, void* d_ws, size_t ws_size,
                              hipStream_t stream) {
  (void)in_sizes; (void)n_in; (void)out_size; (void)ws_size;

  const float* x      = (const float*)d_in[0];
  const float* W_in   = (const float*)d_in[1];
  const float* b_in   = (const float*)d_in[2];
  const float* ln_w   = (const float*)d_in[3];
  const float* ln_b   = (const float*)d_in[4];
  const float* W_ip   = (const float*)d_in[5];
  const float* conv_w = (const float*)d_in[6];
  const float* conv_b = (const float*)d_in[7];
  const float* W_x    = (const float*)d_in[8];
  const float* W_dt   = (const float*)d_in[9];
  const float* b_dt   = (const float*)d_in[10];
  const float* A_log  = (const float*)d_in[11];
  const float* Dp     = (const float*)d_in[12];
  const float* W_out  = (const float*)d_in[13];
  const float* fln_w  = (const float*)d_in[14];
  const float* fln_b  = (const float*)d_in[15];
  const float* W_h1   = (const float*)d_in[16];
  const float* b_h1   = (const float*)d_in[17];
  const float* W_h2   = (const float*)d_in[18];
  const float* b_h2   = (const float*)d_in[19];
  float* out = (float*)d_out;

  char* p = (char*)d_ws;
  auto carve = [&](size_t bytes) -> char* {
    char* r = p; p += (bytes + 255) & ~(size_t)255; return r;
  };
  float*  h_buf   = (float*) carve((size_t)M_ * DM * 4);
  __bf16* hn_bf   = (__bf16*)carve((size_t)M_ * DM * 2);
  __bf16* xzbf    = (__bf16*)carve((size_t)M_ * 2 * DI * 2);
  __bf16* ubf     = (__bf16*)carve((size_t)M_ * DI * 2);
  float*  xdb     = (float*) carve((size_t)M_ * (DTR + 2 * DS) * 4);
  __bf16* dtr_bf  = (__bf16*)carve((size_t)M_ * DTR * 2);
  __bf16* dtbbf   = (__bf16*)carve((size_t)M_ * DI * 2);
  float*  ysb     = (float*) carve((size_t)M_ * DI * 4);
  __bf16* ybf     = (__bf16*)carve((size_t)M_ * DI * 2);
  __bf16* g_bf    = (__bf16*)carve((size_t)M_ * DFF * 2);
  __bf16* wip_bf  = (__bf16*)carve((size_t)NL * 2 * DI * DM * 2);
  __bf16* wx_bf   = (__bf16*)carve((size_t)NL * (DTR + 2 * DS) * DI * 2);
  __bf16* wdt_bf  = (__bf16*)carve((size_t)NL * DI * DTR * 2);
  __bf16* wout_bf = (__bf16*)carve((size_t)NL * DM * DI * 2);
  __bf16* wh1_bf  = (__bf16*)carve((size_t)DFF * DM * 2);
  __bf16* wh2_bf  = (__bf16*)carve((size_t)FD * DFF * 2);
  float*  Aneg    = (float*) carve((size_t)NL * DI * DS * 4);

  auto cvt = [&](const float* src, __bf16* dst, int n) {
    k_cvt_bf16<<<(n + 255) / 256, 256, 0, stream>>>(src, dst, n);
  };
  auto gemm = [&](const __bf16* A, const __bf16* W, const float* bias,
                  float* Cf, __bf16* Cbf, int N, int K, int act, int addC) {
    dim3 g((N + 63) / 64, M_ / 128);
    k_gemm_wmma<<<g, 256, 0, stream>>>(A, W, bias, Cf, Cbf, M_, N, K, act, addC);
  };

  // weight precompute (deterministic every call)
  cvt(W_ip,  wip_bf,  NL * 2 * DI * DM);
  cvt(W_x,   wx_bf,   NL * (DTR + 2 * DS) * DI);
  cvt(W_dt,  wdt_bf,  NL * DI * DTR);
  cvt(W_out, wout_bf, NL * DM * DI);
  cvt(W_h1,  wh1_bf,  DFF * DM);
  cvt(W_h2,  wh2_bf,  FD * DFF);
  k_negexp<<<(NL * DI * DS) / 256, 256, 0, stream>>>(A_log, Aneg, NL * DI * DS);

  // input projection
  k_in_proj<<<(M_ * DM) / 256, 256, 0, stream>>>(x, W_in, b_in, h_buf);

  for (int l = 0; l < NL; ++l) {
    k_layernorm_bf16<<<M_, 256, 0, stream>>>(h_buf, ln_w + l * DM, ln_b + l * DM, hn_bf);
    // in_proj: (M,512) x (2048,512)^T -> xz (bf16)
    gemm(hn_bf, wip_bf + (size_t)l * 2 * DI * DM, nullptr, nullptr, xzbf, 2 * DI, DM, 0, 0);
    // depthwise causal conv + SiLU
    k_conv_silu<<<(M_ * DI) / 256, 256, 0, stream>>>(xzbf, conv_w + l * DI * DC,
                                                     conv_b + l * DI, ubf);
    // x_proj: (M,1024) x (64,1024)^T -> xdb (f32: feeds scan B/C)
    gemm(ubf, wx_bf + (size_t)l * (DTR + 2 * DS) * DI, nullptr, xdb, nullptr,
         DTR + 2 * DS, DI, 0, 0);
    k_slice_dtr<<<(M_ * DTR) / 256, 256, 0, stream>>>(xdb, dtr_bf);
    // dt_proj + softplus: (M,32) x (1024,32)^T -> dt (bf16)
    gemm(dtr_bf, wdt_bf + (size_t)l * DI * DTR, b_dt + l * DI, nullptr, dtbbf, DI, DTR, 3, 0);
    // selective scan
    k_scan<<<(B_ * DI * DS) / 256, 256, 0, stream>>>(dtbbf, ubf, xdb,
                                                     Aneg + l * DI * DS, ysb);
    // gate
    k_gate<<<(M_ * DI) / 256, 256, 0, stream>>>(ysb, ubf, xzbf, Dp + l * DI, ybf);
    // out_proj + residual add into h
    gemm(ybf, wout_bf + (size_t)l * DM * DI, nullptr, h_buf, nullptr, DM, DI, 0, 1);
  }

  // final LN + head: Linear -> GELU -> Linear
  k_layernorm_bf16<<<M_, 256, 0, stream>>>(h_buf, fln_w, fln_b, hn_bf);
  gemm(hn_bf, wh1_bf, b_h1, nullptr, g_bf, DFF, DM, 2, 0);
  gemm(g_bf, wh2_bf, b_h2, out, nullptr, FD, DFF, 0, 0);
}
